// MergedQKVParallelLinearWithDelta_28973849379101
// MI455X (gfx1250) — matrix operand order repaired and checked by reading, not compiled, not run on hardware
//
#include <hip/hip_runtime.h>

// ---------------------------------------------------------------------------
// MergedQKVParallelLinearWithDelta for MI455X (gfx1250, wave32, WMMA).
//
// out = x @ w_base^T + bias + x @ Wd[idx[t]]^T
//   Wd[d][o][k] = (nib(qw) - (nib(qz)+1)) * sc[d][o]      (single GPTQ group)
//
// bf16 split-precision WMMA strategy:
//   base : A_hi*B_hi + A_hi*B_lo + A_lo*B_hi           (3 wmma / tile / kstep)
//   delta: raw nibble GEMM R_d = x * w4^T (w4 exact in bf16), 4 adapters,
//          hi+lo A                                      (8 wmma / tile / kstep)
//   rowsum S[m] = x * ones^T via wmma (acc layout == C) (2 wmma / wave / kstep)
// Epilogue per lane: d = indices[m];
//   out = base + bias[n] + sc[d][n]*R_d - sc[d][n]*(z[d][n]+1)*S[m]
//
// v2: software-pipelined global loads — the x / w_base / qweight chunks for
// K-step kk+1 are loaded into registers right after the barrier, so VMEM is
// in flight underneath the 24-WMMA compute phase (XDL co-execution hides the
// bf16 conversion VALU as well). LDS stays single-buffered (~35KB/block).
// ---------------------------------------------------------------------------

typedef __attribute__((ext_vector_type(16))) __bf16 v16bf;
typedef __attribute__((ext_vector_type(8)))  __bf16 v8bf;
typedef __attribute__((ext_vector_type(8)))  float  v8f;

#define T_TOK   1024
#define IN_DIM  4096
#define OUT_DIM 6144
#define QO      4096
#define KVO     1024
#define BM      64
#define BN      64
#define BK      32
#define NKSTEP  (IN_DIM / BK)
#define KP      (IN_DIM / 8)   // packed int32 words per delta row

__device__ __forceinline__ v16bf join8(v8bf a, v8bf b) {
  return __builtin_shufflevector(a, b, 0,1,2,3,4,5,6,7,8,9,10,11,12,13,14,15);
}

__device__ __forceinline__ v8f wmma_bf16(v16bf a, v16bf b, v8f c) {
  // (neg_a, A, neg_b, B, c_mod, C, reuse_a, reuse_b)
  return __builtin_amdgcn_wmma_f32_16x16x32_bf16(false, a, false, b, (short)0, c,
                                                 false, false);
}

__device__ __forceinline__ v8f vzero8() {
  v8f r;
#pragma unroll
  for (int i = 0; i < 8; ++i) r[i] = 0.0f;
  return r;
}

// f32x8 (two float4 regs) -> bf16 hi/lo split
__device__ __forceinline__ void split8(const float4 f0, const float4 f1,
                                       v8bf& h, v8bf& l) {
  const float vv[8] = {f0.x, f0.y, f0.z, f0.w, f1.x, f1.y, f1.z, f1.w};
#pragma unroll
  for (int i = 0; i < 8; ++i) {
    const __bf16 hb = (__bf16)vv[i];
    h[i] = hb;
    l[i] = (__bf16)(vv[i] - (float)hb);
  }
}

__global__ __launch_bounds__(256, 2)
void qkv_delta_wmma_kernel(
    const float* __restrict__ x, const float* __restrict__ w_base,
    const float* __restrict__ bias,
    const int* __restrict__ qw_q, const int* __restrict__ qw_k,
    const int* __restrict__ qw_v,
    const int* __restrict__ qz_q, const int* __restrict__ qz_k,
    const int* __restrict__ qz_v,
    const float* __restrict__ sc_q, const float* __restrict__ sc_k,
    const float* __restrict__ sc_v,
    const int* __restrict__ indices,
    float* __restrict__ out)
{
  // ---- LDS staging (single-buffered; register pipeline covers latency) ----
  __shared__ __align__(16) __bf16 Ahi[BM][BK];
  __shared__ __align__(16) __bf16 Alo[BM][BK];
  __shared__ __align__(16) __bf16 Bhi[BN][BK];
  __shared__ __align__(16) __bf16 Blo[BN][BK];
  __shared__ __align__(16) __bf16 W4[4][BN][BK];   // dequant nibbles (exact)
  __shared__ float c1LDS[4 * BN];                  // sc[d][n]
  __shared__ float c2LDS[4 * BN];                  // sc[d][n]*(z[d][n]+1)
  __shared__ float biasLDS[BN];
  __shared__ int   idxLDS[BM];

  const int t  = threadIdx.x;
  const int n0 = blockIdx.x * BN;
  const int m0 = blockIdx.y * BM;

  // ---- pick q/k/v stack (uniform per block: BN divides stack boundaries) --
  const int* qw; const int* qz; const float* sc; int O, obase;
  if (n0 < QO)            { qw = qw_q; qz = qz_q; sc = sc_q; O = QO;  obase = n0; }
  else if (n0 < QO + KVO) { qw = qw_k; qz = qz_k; sc = sc_k; O = KVO; obase = n0 - QO; }
  else                    { qw = qw_v; qz = qz_v; sc = sc_v; O = KVO; obase = n0 - QO - KVO; }

  // ---- prologue: epilogue constants to LDS --------------------------------
  if (t < BM) idxLDS[t] = indices[m0 + t];
  if (t < BN) biasLDS[t] = bias[n0 + t];
  {
    const int d = t >> 6;          // 0..3
    const int n = t & 63;          // 0..63
    const int o = obase + n;
    const float s  = sc[(size_t)d * O + o];
    const int   zw = qz[(size_t)d * (O >> 3) + (o >> 3)];
    const float z  = (float)((unsigned)(zw >> ((o & 7) * 4)) & 0xFu);
    c1LDS[t] = s;
    c2LDS[t] = s * (z + 1.0f);
  }

  // ---- per-wave geometry --------------------------------------------------
  const int wv     = t >> 5;
  const int lane   = t & 31;
  const int lr     = lane & 15;        // fragment row within 16
  const int hih    = lane >> 4;        // K-half selector
  const int m_tile = wv >> 1;          // 0..3
  const int ntb    = (wv & 1) * 2;     // first of two N tiles: 0 or 2

  // ---- loader geometry ----------------------------------------------------
  const int la_r = t >> 2;             // 0..63 row
  const int la_c = (t & 3) * 8;        // 0,8,16,24 col chunk
  const int ld_d = t >> 6;             // 0..3 adapter
  const int ld_n = t & 63;             // 0..63 delta row

  const float* xrow = x      + (size_t)(m0 + la_r) * IN_DIM + la_c;
  const float* wrow = w_base + (size_t)(n0 + la_r) * IN_DIM + la_c;
  const int*   qrow = qw + ((size_t)ld_d * O + obase + ld_n) * KP;

  // ---- accumulators -------------------------------------------------------
  v8f accB[2];
  v8f accR[2][4];
  v8f accS = vzero8();
#pragma unroll
  for (int i = 0; i < 2; ++i) {
    accB[i] = vzero8();
#pragma unroll
    for (int d = 0; d < 4; ++d) accR[i][d] = vzero8();
  }

  v16bf onesB;
#pragma unroll
  for (int i = 0; i < 16; ++i) onesB[i] = (__bf16)1.0f;

  // ---- register pipeline: preload K-step 0 --------------------------------
  float4 ra0 = *(const float4*)(xrow);
  float4 ra1 = *(const float4*)(xrow + 4);
  float4 rb0 = *(const float4*)(wrow);
  float4 rb1 = *(const float4*)(wrow + 4);
  int4   rq  = *(const int4*)(qrow);

  // ---- main K loop --------------------------------------------------------
  for (int kk = 0; kk < NKSTEP; ++kk) {
    __syncthreads();   // previous iteration's LDS readers are done

    // stage K-step kk from pipeline registers (conversion co-executes w/ XDL)
    {
      v8bf h, l;
      split8(ra0, ra1, h, l);
      *(v8bf*)&Ahi[la_r][la_c] = h;
      *(v8bf*)&Alo[la_r][la_c] = l;
      split8(rb0, rb1, h, l);
      *(v8bf*)&Bhi[la_r][la_c] = h;
      *(v8bf*)&Blo[la_r][la_c] = l;
      const int wds[4] = {rq.x, rq.y, rq.z, rq.w};
#pragma unroll
      for (int j = 0; j < 4; ++j) {
        v8bf o8;
#pragma unroll
        for (int b = 0; b < 8; ++b)
          o8[b] = (__bf16)(float)((unsigned)(wds[j] >> (4 * b)) & 0xFu);
        *(v8bf*)&W4[ld_d][ld_n][j * 8] = o8;
      }
    }
    __syncthreads();

    // issue next K-step's global loads: VMEM in flight under the WMMA block
    if (kk + 1 < NKSTEP) {
      const int k1 = (kk + 1) * BK;
      ra0 = *(const float4*)(xrow + k1);
      ra1 = *(const float4*)(xrow + k1 + 4);
      rb0 = *(const float4*)(wrow + k1);
      rb1 = *(const float4*)(wrow + k1 + 4);
      rq  = *(const int4*)(qrow + (k1 >> 3));
      // first-touch HBM streaming prefetch two steps ahead (L2-resident after)
      __builtin_prefetch(xrow + k1 + BK, 0, 0);
      __builtin_prefetch(wrow + k1 + BK, 0, 0);
    }

    // A fragments (contiguous K, matching B: lane<16 -> K0..15;
    // lane>=16 -> K16..31)
    const int am = m_tile * 16 + lr;
    const int ac = hih * 16;
    const v16bf ahi = join8(*(v8bf*)&Ahi[am][ac], *(v8bf*)&Ahi[am][ac + 8]);
    const v16bf alo = join8(*(v8bf*)&Alo[am][ac], *(v8bf*)&Alo[am][ac + 8]);

    // row-sum via ones-B (accumulator layout == C layout -> S[m] per lane)
    accS = wmma_bf16(ahi, onesB, accS);
    accS = wmma_bf16(alo, onesB, accS);

#pragma unroll
    for (int nt = 0; nt < 2; ++nt) {
      const int n  = (ntb + nt) * 16 + lr;
      const int ks = hih * 16;  // B layout: lane<16 -> K0..15, lane>=16 -> K16..31
      const v16bf bhi = join8(*(v8bf*)&Bhi[n][ks], *(v8bf*)&Bhi[n][ks + 8]);
      const v16bf blo = join8(*(v8bf*)&Blo[n][ks], *(v8bf*)&Blo[n][ks + 8]);
      accB[nt] = wmma_bf16(ahi, bhi, accB[nt]);
      accB[nt] = wmma_bf16(ahi, blo, accB[nt]);
      accB[nt] = wmma_bf16(alo, bhi, accB[nt]);
#pragma unroll
      for (int d = 0; d < 4; ++d) {
        const v16bf wf = join8(*(v8bf*)&W4[d][n][ks], *(v8bf*)&W4[d][n][ks + 8]);
        accR[nt][d] = wmma_bf16(ahi, wf, accR[nt][d]);
        accR[nt][d] = wmma_bf16(alo, wf, accR[nt][d]);
      }
    }
  }

  // ---- epilogue: per-lane adapter select + scale/zero fold ----------------
#pragma unroll
  for (int nt = 0; nt < 2; ++nt) {
#pragma unroll
    for (int e = 0; e < 8; ++e) {
      const int ml = m_tile * 16 + e + hih * 8;    // C layout: M = e (+8 hi half)
      const int nl = (ntb + nt) * 16 + lr;         // C layout: N = lane%16
      const int d  = idxLDS[ml];
      float R = accR[nt][0][e];
      R = (d == 1) ? accR[nt][1][e] : R;
      R = (d == 2) ? accR[nt][2][e] : R;
      R = (d == 3) ? accR[nt][3][e] : R;
      const float v = accB[nt][e] + biasLDS[nl]
                    + c1LDS[d * 64 + nl] * R
                    - c2LDS[d * 64 + nl] * accS[e];
      out[(size_t)(m0 + ml) * OUT_DIM + n0 + nl] = v;
    }
  }
}

extern "C" void kernel_launch(void* const* d_in, const int* in_sizes, int n_in,
                              void* d_out, int out_size, void* d_ws, size_t ws_size,
                              hipStream_t stream) {
  (void)in_sizes; (void)n_in; (void)out_size; (void)d_ws; (void)ws_size;
  const float* x      = (const float*)d_in[0];
  const float* w_base = (const float*)d_in[1];
  const float* bias   = (const float*)d_in[2];
  const int*   qw_q   = (const int*)d_in[3];
  const int*   qw_k   = (const int*)d_in[4];
  const int*   qw_v   = (const int*)d_in[5];
  const int*   qz_q   = (const int*)d_in[6];
  const int*   qz_k   = (const int*)d_in[7];
  const int*   qz_v   = (const int*)d_in[8];
  const float* sc_q   = (const float*)d_in[9];
  const float* sc_k   = (const float*)d_in[10];
  const float* sc_v   = (const float*)d_in[11];
  const int*   idx    = (const int*)d_in[12];
  float*       outp   = (float*)d_out;

  dim3 grid(OUT_DIM / BN, T_TOK / BM);
  qkv_delta_wmma_kernel<<<grid, 256, 0, stream>>>(
      x, w_base, bias, qw_q, qw_k, qw_v, qz_q, qz_k, qz_v,
      sc_q, sc_k, sc_v, idx, outp);
}
